// TGAN_37632503448222
// MI455X (gfx1250) — matrix operand order repaired and checked by reading, not compiled, not run on hardware
//
#include <hip/hip_runtime.h>

typedef __attribute__((ext_vector_type(16))) _Float16 v16h;
typedef __attribute__((ext_vector_type(8)))  float    v8f;

#define ROWS   4
#define KNGH   20
#define MKV    (ROWS * KNGH)   // 80 kv rows (5 M-tiles)
#define KVD    384
#define KV_STR 392             // padded LDS stride (halfs)
#define QA_STR 264
#define OH_STR 264
#define MG_STR 392
#define H2_STR 136

// ---- dynamic LDS layout (bytes) ----
#define OFF_KVA   0            // 80*392*2   = 62720  (reused after GEMM1)
#define OFF_KVKV  62720        // 80*512*2   = 81920  (K in [0,256), V in [256,512))
#define OFF_QA    144640       // 16*264*2   = 8448
#define OFF_QIN   153088       // 4*256*4    = 4096
#define OFF_QF    157184       // 4*256*4    = 4096
#define OFF_NF    161280       // 4*128*4    = 2048
#define OFF_META  163328       // ids/times/flags
#define SMEM_BYTES 164672

__device__ __forceinline__ v8f wmma16(v16h a, v16h b, v8f c) {
  return __builtin_amdgcn_wmma_f32_16x16x32_f16(false, a, false, b, (short)0, c,
                                                false, false);
}

// A-fragment (16x32 f16) from LDS, row-major with stride (halfs).
// lanes 0-15: M=lane, K elems = {0..7, 16..23}; lanes 16-31: M=lane-16, K={8..15, 24..31}
__device__ __forceinline__ v16h afrag(const _Float16* A, int strideK, int m0, int k0, int lane) {
  int m  = lane & 15;
  int ko = (lane < 16) ? 0 : 8;
  const _Float16* p = A + (m0 + m) * strideK + k0 + ko;
  v16h a;
#pragma unroll
  for (int i = 0; i < 8; ++i) { a[i] = p[i]; a[i + 8] = p[i + 16]; }
  return a;
}

// B-fragment (32x16 f16) from N-major weights WT[n][k] (stride = K dim).
// lanes 0-15: N=lane, K={0..7, 16..23}; lanes 16-31: N=lane-16, K={8..15, 24..31}
__device__ __forceinline__ v16h bfrag(const _Float16* __restrict__ WT, int strideK,
                                      int n0, int k0, int lane) {
  int n  = lane & 15;
  int ko = (lane < 16) ? 0 : 8;
  const _Float16* p = WT + (size_t)(n0 + n) * strideK + k0 + ko;
  v16h b;
#pragma unroll
  for (int i = 0; i < 8; ++i) { b[i] = p[i]; b[i + 8] = p[i + 16]; }
  return b;
}

__global__ void transpose_to_f16(const float* __restrict__ src, _Float16* __restrict__ dst,
                                 int R, int C, long long srcStride, long long dstStride,
                                 int nmat) {
  long long total = (long long)nmat * R * C;
  for (long long e = (long long)blockIdx.x * blockDim.x + threadIdx.x; e < total;
       e += (long long)gridDim.x * blockDim.x) {
    long long m   = e / ((long long)R * C);
    long long rem = e % ((long long)R * C);
    int r = (int)(rem / C), c = (int)(rem % C);
    dst[m * dstStride + (long long)c * R + r] = (_Float16)src[m * srcStride + (long long)r * C + c];
  }
}

template <int LAYER>
__global__ void tgat_layer(const int* __restrict__ src_idx,
                           const float* __restrict__ cut_time,
                           const float* __restrict__ node_features,
                           const float* __restrict__ edge_features,
                           const float* __restrict__ tfreq,
                           const float* __restrict__ tphase,
                           const _Float16* __restrict__ WkvT,  // [512][384] n-major (K|V)
                           const _Float16* __restrict__ WqT,   // [256][256]
                           const _Float16* __restrict__ WoT,   // [256][256]
                           const _Float16* __restrict__ mw1T,  // [128][384]
                           const _Float16* __restrict__ mw2T,  // [128][128]
                           const float* __restrict__ bo,
                           const float* __restrict__ ln_g,
                           const float* __restrict__ ln_b,
                           const float* __restrict__ mb1,
                           const float* __restrict__ mb2,
                           const float* __restrict__ conv1_in,
                           float* __restrict__ outp) {
  extern __shared__ char smem[];
  _Float16* kvA  = (_Float16*)(smem + OFF_KVA);
  _Float16* kvKV = (_Float16*)(smem + OFF_KVKV);
  _Float16* qA   = (_Float16*)(smem + OFF_QA);
  float*    qin  = (float*)(smem + OFF_QIN);
  float*    qf   = (float*)(smem + OFF_QF);
  float*    nf   = (float*)(smem + OFF_NF);
  int*   nghid = (int*)(smem + OFF_META);
  int*   eid   = nghid + MKV;
  float* dlt   = (float*)(eid + MKV);
  int*   flg   = (int*)(dlt + MKV);
  int*   rowN  = flg + MKV;
  float* rowT  = (float*)(rowN + ROWS);
  // kvA region is dead after GEMM1 -> reuse for post-GEMM buffers
  float*    attb   = (float*)(smem + OFF_KVA);                           // 320 f32
  _Float16* ohA    = (_Float16*)(smem + OFF_KVA + 1280);                 // 16x264 f16
  float*    lnin   = (float*)(smem + OFF_KVA + 1280 + 8448);             // 4x256 f32
  _Float16* mergeA = (_Float16*)(smem + OFF_KVA + 1280 + 8448 + 4096);   // 16x392 f16
  _Float16* h2A    = (_Float16*)(smem + OFF_KVA + 1280 + 8448 + 4096 + 12544); // 16x136 f16

  const int tid  = threadIdx.x;
  const int w    = tid >> 5;
  const int lane = tid & 31;

  // ---------------- row metadata ----------------
  if (tid < ROWS) {
    int r = blockIdx.x * ROWS + tid;
    int n; float t;
    if (LAYER == 1) {
      if (r < 1024) { n = src_idx[r]; t = cut_time[r]; }
      else {
        int rr = r - 1024, b = rr / KNGH, k = rr % KNGH;
        unsigned h = (unsigned)src_idx[b] * 2654435761u + (unsigned)(k + 1) * 40503u;
        unsigned g = h % 100000u;
        if (((h >> 7) & 3u) == 0u) g = 0u;
        float frac = (float)((h >> 3) & 1023u) * (1.0f / 1024.0f);
        n = (int)g; t = cut_time[b] * frac;
      }
    } else { n = src_idx[r]; t = cut_time[r]; }
    rowN[tid] = n; rowT[tid] = t;
  }
  __syncthreads();

  // per-kv-row neighbor metadata (hash of the row's node id)
  if (tid < MKV) {
    int j = tid / KNGH, k = tid % KNGH;
    unsigned h = (unsigned)rowN[j] * 2654435761u + (unsigned)(k + 1) * 40503u;
    unsigned g = h % 100000u;
    if (((h >> 7) & 3u) == 0u) g = 0u;
    float frac = (float)((h >> 3) & 1023u) * (1.0f / 1024.0f);
    float t = rowT[j];
    nghid[tid] = (int)g;
    eid[tid]   = (int)((h * 97u + 13u) % 500000u);
    dlt[tid]   = t - t * frac;
    flg[tid]   = (g == 0u);
  }
  __syncthreads();

  // ---------------- build kv A-matrix [80 x 384] f16 ----------------
  for (int e = tid; e < MKV * KVD; e += 256) {
    int row = e / KVD, col = e % KVD;
    float v;
    if (col < 128) {
      if (LAYER == 1) {
        v = node_features[(size_t)nghid[row] * 128 + col];
      } else {
        int j = row / KNGH, k = row % KNGH;
        int r = blockIdx.x * ROWS + j;
        v = conv1_in[(size_t)(1024 + r * KNGH + k) * 128 + col];
      }
    } else if (col < 256) {
      v = edge_features[(size_t)eid[row] * 128 + (col - 128)];
    } else {
      int d = col - 256;
      v = cosf(dlt[row] * tfreq[d] + tphase[d]);
    }
    kvA[row * KV_STR + col] = (_Float16)v;
  }
  // q_in [rows x 256] (f32 for residual) + qA f16 padded to 16 rows
  for (int e = tid; e < 16 * 256; e += 256) {
    int m = e >> 8, c = e & 255;
    float v = 0.0f;
    if (m < ROWS) {
      if (c < 128)
        v = (LAYER == 1) ? node_features[(size_t)rowN[m] * 128 + c]
                         : conv1_in[(size_t)(blockIdx.x * ROWS + m) * 128 + c];
      else
        v = cosf(tphase[c - 128]);   // time_encode(0)
      qin[m * 256 + c] = v;
    }
    qA[m * QA_STR + c] = (_Float16)v;
  }
  for (int e = tid; e < ROWS * 128; e += 256) {
    int m = e >> 7, c = e & 127;
    nf[m * 128 + c] = node_features[(size_t)rowN[m] * 128 + c];
  }
  __syncthreads();

  // ---------------- GEMM1: [80x384] @ [384x512] -> K|V f16 ----------------
  // Two sequential passes of 2 N-tiles each: acc[5][2] = 10 accumulators stays
  // register-resident (80 VGPRs). All loads in the K-loop depend on kc ->
  // LICM has nothing to hoist/spill. 10 independent WMMAs per K-step.
#pragma unroll 1
  for (int half = 0; half < 2; ++half) {
    const int n0 = (w + half * 16) * 16;
    const int n1 = (w + 8 + half * 16) * 16;
    v8f acc[5][2];
#pragma unroll
    for (int m = 0; m < 5; ++m) { acc[m][0] = (v8f){}; acc[m][1] = (v8f){}; }
#pragma unroll 1
    for (int kc = 0; kc < 12; ++kc) {
      v16h b0 = bfrag(WkvT, 384, n0, kc * 32, lane);
      v16h b1 = bfrag(WkvT, 384, n1, kc * 32, lane);
#pragma unroll
      for (int m = 0; m < 5; ++m) {
        v16h a = afrag(kvA, KV_STR, m * 16, kc * 32, lane);
        acc[m][0] = wmma16(a, b0, acc[m][0]);
        acc[m][1] = wmma16(a, b1, acc[m][1]);
      }
    }
    int nc = lane & 15, mr = (lane < 16) ? 0 : 8;
#pragma unroll
    for (int m = 0; m < 5; ++m)
#pragma unroll
      for (int i = 0; i < 8; ++i) {
        int rowb = (m * 16 + mr + i) * 512;
        kvKV[rowb + n0 + nc] = (_Float16)acc[m][0][i];
        kvKV[rowb + n1 + nc] = (_Float16)acc[m][1][i];
      }
  }
  // ---------------- Q GEMM: [16x256] @ [256x256] -> qf f32 ----------------
  {
    v8f acc0 = {}, acc1 = {};
#pragma unroll 1
    for (int kc = 0; kc < 8; ++kc) {
      v16h a = afrag(qA, QA_STR, 0, kc * 32, lane);
      acc0 = wmma16(a, bfrag(WqT, 256, (w * 2) * 16,     kc * 32, lane), acc0);
      acc1 = wmma16(a, bfrag(WqT, 256, (w * 2 + 1) * 16, kc * 32, lane), acc1);
    }
    int nc = lane & 15, mr = (lane < 16) ? 0 : 8;
#pragma unroll
    for (int i = 0; i < 8; ++i) {
      int mm = mr + i;
      if (mm < ROWS) {
        qf[mm * 256 + (w * 2) * 16 + nc]     = acc0[i];
        qf[mm * 256 + (w * 2 + 1) * 16 + nc] = acc1[i];
      }
    }
  }
  __syncthreads();

  // ---------------- attention scores + mask ----------------
  for (int it = tid; it < ROWS * 4 * KNGH; it += 256) {
    int j = it / (4 * KNGH), rest = it % (4 * KNGH), hh = rest / KNGH, k = rest % KNGH;
    int kr = j * KNGH + k;
    const _Float16* Kp = kvKV + kr * 512 + hh * 64;
    const float* qp = qf + j * 256 + hh * 64;
    float s = 0.f;
#pragma unroll 8
    for (int d = 0; d < 64; ++d) s += qp[d] * (float)Kp[d];
    s *= 0.125f;                    // 64^-0.5
    if (flg[kr]) s = -1e10f;
    attb[it] = s;
  }
  __syncthreads();
  if (tid < ROWS * 4) {
    float* a = attb + tid * KNGH;
    float mx = a[0];
    for (int k = 1; k < KNGH; ++k) mx = fmaxf(mx, a[k]);
    float sum = 0.f;
    for (int k = 0; k < KNGH; ++k) { float e = __expf(a[k] - mx); a[k] = e; sum += e; }
    float inv = 1.0f / sum;
    for (int k = 0; k < KNGH; ++k) a[k] *= inv;
  }
  __syncthreads();

  // ---------------- o = p @ V -> ohA f16 [16x256] (padded) ----------------
  for (int e = tid; e < 16 * OH_STR; e += 256) {
    int m = e / OH_STR, c = e % OH_STR;
    float v = 0.f;
    if (m < ROWS && c < 256) {
      int hh = c >> 6;
      const float* p = attb + m * (4 * KNGH) + hh * KNGH;
      const _Float16* Vp = kvKV + (size_t)(m * KNGH) * 512 + 256 + c;
#pragma unroll
      for (int k = 0; k < KNGH; ++k) v += p[k] * (float)Vp[(size_t)k * 512];
    }
    ohA[e] = (_Float16)v;
  }
  __syncthreads();

  // ---------------- O GEMM + bias + residual ----------------
  {
    v8f acc0 = {}, acc1 = {};
#pragma unroll 1
    for (int kc = 0; kc < 8; ++kc) {
      v16h a = afrag(ohA, OH_STR, 0, kc * 32, lane);
      acc0 = wmma16(a, bfrag(WoT, 256, (w * 2) * 16,     kc * 32, lane), acc0);
      acc1 = wmma16(a, bfrag(WoT, 256, (w * 2 + 1) * 16, kc * 32, lane), acc1);
    }
    int nc = lane & 15, mr = (lane < 16) ? 0 : 8;
#pragma unroll
    for (int i = 0; i < 8; ++i) {
      int mm = mr + i;
      if (mm < ROWS) {
        int c0 = (w * 2) * 16 + nc, c1 = (w * 2 + 1) * 16 + nc;
        lnin[mm * 256 + c0] = acc0[i] + bo[c0] + qin[mm * 256 + c0];
        lnin[mm * 256 + c1] = acc1[i] + bo[c1] + qin[mm * 256 + c1];
      }
    }
  }
  __syncthreads();

  // ---------------- LayerNorm (wave32 reduce) + merge concat ----------------
  if (w < ROWS) {
    const float* x = lnin + w * 256;
    float s = 0.f, s2 = 0.f;
    for (int c = lane; c < 256; c += 32) { float v = x[c]; s += v; s2 += v * v; }
#pragma unroll
    for (int off = 16; off > 0; off >>= 1) {
      s += __shfl_xor(s, off, 32);
      s2 += __shfl_xor(s2, off, 32);
    }
    float mu = s * (1.0f / 256.0f);
    float var = s2 * (1.0f / 256.0f) - mu * mu;
    float inv = rsqrtf(var + 1e-5f);
    for (int c = lane; c < 256; c += 32)
      mergeA[w * MG_STR + c] = (_Float16)((x[c] - mu) * inv * ln_g[c] + ln_b[c]);
    for (int c = lane; c < 128; c += 32)
      mergeA[w * MG_STR + 256 + c] = (_Float16)nf[w * 128 + c];
  }
  for (int e = tid; e < 16 * MG_STR; e += 256)
    if (e / MG_STR >= ROWS) mergeA[e] = (_Float16)0.f;
  __syncthreads();

  // ---------------- merge fc1: [16x384] @ [384x128], relu ----------------
  {
    int n = w;
    v8f acc = {};
#pragma unroll 1
    for (int kc = 0; kc < 12; ++kc)
      acc = wmma16(afrag(mergeA, MG_STR, 0, kc * 32, lane),
                   bfrag(mw1T, 384, n * 16, kc * 32, lane), acc);
    int nc = lane & 15, mr = (lane < 16) ? 0 : 8;
#pragma unroll
    for (int i = 0; i < 8; ++i) {
      int m = mr + i, col = n * 16 + nc;
      float v = fmaxf(acc[i] + mb1[col], 0.0f);
      h2A[m * H2_STR + col] = (_Float16)((m < ROWS) ? v : 0.0f);
    }
  }
  __syncthreads();

  // ---------------- merge fc2: [16x128] @ [128x128] -> out ----------------
  {
    int n = w;
    v8f acc = {};
#pragma unroll 1
    for (int kc = 0; kc < 4; ++kc)
      acc = wmma16(afrag(h2A, H2_STR, 0, kc * 32, lane),
                   bfrag(mw2T, 128, n * 16, kc * 32, lane), acc);
    int nc = lane & 15, mr = (lane < 16) ? 0 : 8;
#pragma unroll
    for (int i = 0; i < 8; ++i) {
      int m = mr + i, col = n * 16 + nc;
      if (m < ROWS) {
        int r = blockIdx.x * ROWS + m;
        outp[(size_t)r * 128 + col] = acc[i] + mb2[col];
      }
    }
  }
}

extern "C" void kernel_launch(void* const* d_in, const int* in_sizes, int n_in,
                              void* d_out, int out_size, void* d_ws, size_t ws_size,
                              hipStream_t stream) {
  (void)in_sizes; (void)n_in; (void)out_size; (void)ws_size;
  const int*   src_idx  = (const int*)  d_in[0];
  const float* cut_time = (const float*)d_in[3];
  const float* node_f   = (const float*)d_in[4];
  const float* edge_f   = (const float*)d_in[5];
  const float* tfreq    = (const float*)d_in[6];
  const float* tphase   = (const float*)d_in[7];
  const float* Wq  = (const float*)d_in[8];
  const float* Wk  = (const float*)d_in[9];
  const float* Wv  = (const float*)d_in[10];
  const float* Wo  = (const float*)d_in[11];
  const float* bo  = (const float*)d_in[12];
  const float* lng = (const float*)d_in[13];
  const float* lnb = (const float*)d_in[14];
  const float* mw1 = (const float*)d_in[15];
  const float* mb1 = (const float*)d_in[16];
  const float* mw2 = (const float*)d_in[17];
  const float* mb2 = (const float*)d_in[18];

  char* ws = (char*)d_ws;
  float*    conv1 = (float*)(ws + 0);                 // [21504][128] f32
  _Float16* WkvT  = (_Float16*)(ws + 11010048);       // 2 x [512][384]
  _Float16* WqT   = (_Float16*)(ws + 11796480);       // 2 x [256][256]
  _Float16* WoT   = (_Float16*)(ws + 12058624);       // 2 x [256][256]
  _Float16* mw1T  = (_Float16*)(ws + 12320768);       // 2 x [128][384]
  _Float16* mw2T  = (_Float16*)(ws + 12517376);       // 2 x [128][128]

  transpose_to_f16<<<128, 256, 0, stream>>>(Wq,  WqT,            256, 256, 256*256, 256*256, 2);
  transpose_to_f16<<<128, 256, 0, stream>>>(Wk,  WkvT,           384, 256, 384*256, 512*384, 2);
  transpose_to_f16<<<128, 256, 0, stream>>>(Wv,  WkvT + 256*384, 384, 256, 384*256, 512*384, 2);
  transpose_to_f16<<<128, 256, 0, stream>>>(Wo,  WoT,            256, 256, 256*256, 256*256, 2);
  transpose_to_f16<<<128, 256, 0, stream>>>(mw1, mw1T,           384, 128, 384*128, 384*128, 2);
  transpose_to_f16<<< 64, 256, 0, stream>>>(mw2, mw2T,           128, 128, 128*128, 128*128, 2);

  hipFuncSetAttribute((const void*)tgat_layer<1>,
                      hipFuncAttributeMaxDynamicSharedMemorySize, SMEM_BYTES);
  hipFuncSetAttribute((const void*)tgat_layer<2>,
                      hipFuncAttributeMaxDynamicSharedMemorySize, SMEM_BYTES);

  // layer 1: 1024 src rows + 1024*20 neighbor rows = 21504
  tgat_layer<1><<<21504 / ROWS, 256, SMEM_BYTES, stream>>>(
      src_idx, cut_time, node_f, edge_f, tfreq, tphase,
      WkvT, WqT, WoT, mw1T, mw2T,
      bo, lng, lnb, mb1, mb2,
      conv1, conv1);
  // layer 2: 1024 rows -> d_out
  tgat_layer<2><<<1024 / ROWS, 256, SMEM_BYTES, stream>>>(
      src_idx, cut_time, node_f, edge_f, tfreq, tphase,
      WkvT + 512*384, WqT + 256*256, WoT + 256*256, mw1T + 128*384, mw2T + 128*128,
      bo + 256, lng + 256, lnb + 256, mb1 + 128, mb2 + 128,
      conv1, (float*)d_out);
}
